// DeformConv2_47115791237775
// MI455X (gfx1250) — matrix operand order repaired and checked by reading, not compiled, not run on hardware
//
#include <hip/hip_runtime.h>
#include <hip/hip_bf16.h>

// DCNv2 forward for MI455X (gfx1250, wave32, WMMA).
//   N=4, Cin=Cout=64, H=W=128, K=3, stride=1, pad=1, dil=1 -> Ho=Wo=128.
// Phase 1: 27-ch offset/mask conv in fp32 (feeds sample coordinates -> keep precise).
// Phase 2+3 fused: bilinear sample * sigmoid(mask) -> f16 LDS panel -> WMMA GEMM
//   out[64,pix] = Wf16[64,576] x B[576,pix], fp32 accum, +bias, ReLU.

#define CIN   64
#define COUT  64
#define KK2   9
#define KDIM  576          // CIN * K2
#define HH    128
#define WW    128
#define NPIC  4
#define HW    (HH * WW)
#define PIXT  16           // pixels per workgroup (one row segment; 128 % 16 == 0)
#define BSTR  584          // LDS halves per pixel column (576 + 8 pad; 584*2 B is 16B-aligned)

typedef __attribute__((ext_vector_type(16))) _Float16 v16h;
typedef __attribute__((ext_vector_type(8)))  _Float16 v8h;
typedef __attribute__((ext_vector_type(8)))  float    v8f;

// ---------------- kernel 0: weights fp32 -> f16, flat [o][c*9 + kh*3 + kw] ----
__global__ __launch_bounds__(256) void dcn_cvtw(const float* __restrict__ w,
                                                _Float16* __restrict__ wf) {
  int i = blockIdx.x * 256 + threadIdx.x;
  if (i < COUT * KDIM) wf[i] = (_Float16)w[i];   // OIHW flat == [o][c*9+k2]
}

// ---------------- kernel 1: 27-channel offset/mask conv, fp32 direct ---------
__global__ __launch_bounds__(256) void dcn_offconv(const float* __restrict__ x,
                                                   const float* __restrict__ w_off,
                                                   const float* __restrict__ b_off,
                                                   float* __restrict__ om) {
  int idx = blockIdx.x * 256 + threadIdx.x;        // [n][oc][ho][wo]
  if (idx >= NPIC * 27 * HW) return;
  int wo = idx & (WW - 1);
  int ho = (idx >> 7) & (HH - 1);
  int oc = (idx >> 14) % 27;
  int n  = idx / (27 * HW);

  const float* xn   = x + (size_t)n * CIN * HW;
  const float* wrow = w_off + (size_t)oc * CIN * KK2;
  float acc = b_off[oc];
  for (int c = 0; c < CIN; ++c) {
    const float* xc = xn + c * HW;
    const float* wc = wrow + c * KK2;
#pragma unroll
    for (int kh = 0; kh < 3; ++kh) {
      int y = ho + kh - 1;
      if ((unsigned)y >= (unsigned)HH) continue;
#pragma unroll
      for (int kw = 0; kw < 3; ++kw) {
        int xx = wo + kw - 1;
        if ((unsigned)xx >= (unsigned)WW) continue;
        acc += xc[y * WW + xx] * wc[kh * 3 + kw];
      }
    }
  }
  om[idx] = acc;
}

// ---------------- kernel 2: fused sample -> LDS f16 -> WMMA GEMM -------------
__global__ __launch_bounds__(128) void dcn_main(const float* __restrict__ x,
                                                const float* __restrict__ om,
                                                const _Float16* __restrict__ wf,
                                                const float* __restrict__ bias,
                                                float* __restrict__ out) {
  __shared__ __attribute__((aligned(32))) _Float16 Bs[PIXT * BSTR];

  int tile = blockIdx.x;               // 4096 tiles total
  int n    = tile >> 10;               // 1024 tiles per image
  int t    = tile & 1023;
  int pixbase = t * PIXT;              // flat ho*128 + wo
  int ho   = pixbase >> 7;
  int wo0  = pixbase & (WW - 1);

  const float* xn  = x  + (size_t)n * CIN * HW;
  const float* omn = om + (size_t)n * 27 * HW;

  int tid = threadIdx.x;

  // ---- stage B panel: sampled[c*9+k2][pix] * sigmoid(mask), as f16 ----------
  // 144 (k2,pix) pairs; each pair shares coords/weights across all 64 channels.
  for (int pair = tid; pair < KK2 * PIXT; pair += 128) {
    int k2  = pair % KK2;
    int pix = pair / KK2;
    int wo  = wo0 + pix;
    int hw  = ho * WW + wo;

    // om channels: dy = 2*k2, dx = 2*k2+1, mask = 18+k2 (see reshape analysis)
    float dy   = omn[(2 * k2)     * HW + hw];
    float dx   = omn[(2 * k2 + 1) * HW + hw];
    float mval = omn[(18 + k2)    * HW + hw];
    float msk  = 1.0f / (1.0f + __expf(-mval));

    float py = (float)(ho - 1 + (k2 / 3)) + dy;
    float px = (float)(wo - 1 + (k2 % 3)) + dx;
    float y0f = floorf(py), x0f = floorf(px);
    int y0 = (int)y0f, x0 = (int)x0f;
    int y1 = y0 + 1,   x1 = x0 + 1;
    float ly = py - y0f, lx = px - x0f;
    float hy = 1.0f - ly, hx = 1.0f - lx;

    float vy0 = ((unsigned)y0 < (unsigned)HH) ? 1.0f : 0.0f;
    float vy1 = ((unsigned)y1 < (unsigned)HH) ? 1.0f : 0.0f;
    float vx0 = ((unsigned)x0 < (unsigned)WW) ? 1.0f : 0.0f;
    float vx1 = ((unsigned)x1 < (unsigned)WW) ? 1.0f : 0.0f;

    float w00 = hy * hx * vy0 * vx0 * msk;
    float w01 = hy * lx * vy0 * vx1 * msk;
    float w10 = ly * hx * vy1 * vx0 * msk;
    float w11 = ly * lx * vy1 * vx1 * msk;

    int yc0 = min(max(y0, 0), HH - 1), yc1 = min(max(y1, 0), HH - 1);
    int xc0 = min(max(x0, 0), WW - 1), xc1 = min(max(x1, 0), WW - 1);
    int i00 = yc0 * WW + xc0, i01 = yc0 * WW + xc1;
    int i10 = yc1 * WW + xc0, i11 = yc1 * WW + xc1;

    _Float16* dst = &Bs[pix * BSTR + k2];
    for (int c = 0; c < CIN; ++c) {
      const float* xc = xn + c * HW;
      float v = w00 * xc[i00] + w01 * xc[i01] + w10 * xc[i10] + w11 * xc[i11];
      dst[c * KK2] = (_Float16)v;      // K index = c*9 + k2 (matches wf layout)
    }
  }
  __syncthreads();

  // ---- WMMA: each of 4 waves owns a 16(Cout) x 16(pix) tile -----------------
  int wave  = tid >> 5;
  int lane  = tid & 31;
  int m     = lane & 15;               // A: row within tile / B,D: column (pixel)
  int hi    = lane >> 4;               // K-half / row+8 selector
  int crow0 = wave * 16;               // Cout base row for this wave

  v8f acc = {};
  for (int kb = 0; kb < KDIM; kb += 32) {
    // A fragment (16-bit 16x32 layout): lanes 0-15 hold K {kb..+7, kb+16..+23},
    // lanes 16-31 hold K {kb+8..+15, kb+24..+31}, all for row crow0+m.
    const _Float16* arow = wf + (size_t)(crow0 + m) * KDIM + kb + hi * 8;
    union { v16h v; v8h h[2]; } au;
    au.h[0] = *(const v8h*)(arow);         // 16B-aligned
    au.h[1] = *(const v8h*)(arow + 16);

    // B fragment (32x16): column = m; same K order as A (lanes 0-15 K {kb..+7, kb+16..+23}, lanes 16-31 +8).
    const _Float16* bcol = &Bs[m * BSTR + kb + hi * 8];
    union { v16h v; v8h h[2]; } bu;
    bu.h[0] = *(const v8h*)(bcol);
    bu.h[1] = *(const v8h*)(bcol + 16);

    acc = __builtin_amdgcn_wmma_f32_16x16x32_f16(
        /*neg_a=*/false, au.v, /*neg_b=*/false, bu.v,
        /*c_mod=*/(short)0, acc, /*reuse_a=*/false, /*reuse_b=*/false);
  }

  // ---- epilogue: D layout -> bias + ReLU + store ----------------------------
  // acc[r] lives at row (crow0 + r + 8*hi), pixel column (wo0 + m).
#pragma unroll
  for (int r = 0; r < 8; ++r) {
    int mo = crow0 + r + 8 * hi;
    float v = acc[r] + bias[mo];
    out[((size_t)n * COUT + mo) * HW + ho * WW + wo0 + m] = fmaxf(v, 0.0f);
  }
}

// ---------------- launcher ---------------------------------------------------
extern "C" void kernel_launch(void* const* d_in, const int* in_sizes, int n_in,
                              void* d_out, int out_size, void* d_ws, size_t ws_size,
                              hipStream_t stream) {
  const float* x     = (const float*)d_in[0];   // [4,64,128,128]
  const float* w_off = (const float*)d_in[1];   // [27,64,3,3]
  const float* b_off = (const float*)d_in[2];   // [27]
  const float* w     = (const float*)d_in[3];   // [64,64,3,3]
  const float* b     = (const float*)d_in[4];   // [64]
  float* out = (float*)d_out;                   // [4,64,128,128]

  // Workspace layout (needs ~7.15 MB): om fp32 then f16 weights (256B aligned).
  char* ws = (char*)d_ws;
  float*    om = (float*)ws;                              // 4*27*16384*4 = 7,077,888 B
  _Float16* wf = (_Float16*)(ws + 7077888);               // 64*576*2     =    73,728 B

  dcn_cvtw   <<<(COUT * KDIM + 255) / 256, 256, 0, stream>>>(w, wf);
  dcn_offconv<<<(NPIC * 27 * HW + 255) / 256, 256, 0, stream>>>(x, w_off, b_off, om);
  dcn_main   <<<NPIC * (HW / PIXT), 128, 0, stream>>>(x, om, wf, b, out);
}
